// SpatioDynamicGCN_14843406975568
// MI455X (gfx1250) — matrix operand order repaired and checked by reading, not compiled, not run on hardware
//
#include <hip/hip_runtime.h>
#include <hip/hip_bf16.h>

// ---------------------------------------------------------------------------
// SpatioDynamicGCN for MI455X (gfx1250): bf16 WMMA fused chain.
// out[g,b] = elu( D_invA @ elu( mu[g,b] @ T1[g] ) @ T2[g] )
// E=64, F=128, B=1024, bands=5. D_invA from squeeze-excite on A.
// 512 threads = 16 wave32 per block; 2 output tiles per wave per stage.
// ---------------------------------------------------------------------------

typedef __attribute__((ext_vector_type(16))) __bf16 v16bf;
typedef __attribute__((ext_vector_type(8)))  __bf16 v8bf;
typedef __attribute__((ext_vector_type(4)))  __bf16 v4bf;
typedef __attribute__((ext_vector_type(8)))  float  v8f;
typedef __attribute__((ext_vector_type(4)))  float  float4v;
typedef __attribute__((ext_vector_type(4)))  unsigned int   uint4v;
typedef __attribute__((ext_vector_type(4)))  unsigned short ushort4v;

#define NBANDS 5
#define NBATCH 1024
#define EDIM   64
#define FDIM   128
#define NB     8            // batch items per workgroup
#define NTHR   512          // 16 wave32

// LDS strides in halves (rows 16B-aligned so B-frags can ds_load_b128)
#define ST_T   136          // T1t / T2t : 128 x 128 (+pad)
#define ST_D   72           // D_invA    : 64 x 64   (+pad)
#define ST_MU  136          // mu tile   : 64 x 128  (+pad)
#define ST_H1  72           // h1^T      : 128 x 64  (+pad)
#define ST_H2  136          // h2        : 64 x 128  (+pad)

#define LDS_HALVES (128*ST_T + 128*ST_T + 64*ST_D + 64*ST_MU + 128*ST_H1 + 64*ST_H2)
#define LDS_BYTES  (LDS_HALVES * 2)

__device__ __forceinline__ unsigned short f2bf(float x) {
    union { __bf16 h; unsigned short u; } cv;
    cv.h = (__bf16)x;                         // fptrunc, RNE
    return cv.u;
}

__device__ __forceinline__ float elu1(float x) {
    return x > 0.f ? x : (__expf(x) - 1.f);
}

// A-matrix 16x32 bf16 fragment from row-major LDS (halves, stride in halves).
// Layout used here (contiguous k, matches load_frag_b): lanes 0-15 hold M=lane%16; lane group g in {0,1}:
//   VGPR j (0..7): K pair = k0 + 16*g + 2*j (+0,+1)
__device__ __forceinline__ v16bf load_frag_a(const unsigned short* lds, int stride,
                                             int row0, int k0, int lane) {
    union { v16bf v; unsigned int u[8]; } f;
    const int m = lane & 15;
    const int g = lane >> 4;
    const unsigned short* rp = lds + (row0 + m) * stride;
#pragma unroll
    for (int j = 0; j < 8; ++j) {
        int K = k0 + (g << 4) + (j << 1);
        f.u[j] = *(const unsigned int*)(rp + K);
    }
    return f.v;
}

// B-matrix 32x16 bf16 fragment from LDS holding B transposed: Bt[n][k].
// Layout used here (contiguous k, matches load_frag_a): lane holds column N = n0 + lane%16; lane group g:
//   16 consecutive K values k0+16*g .. k0+16*g+15  -> two b128 LDS loads.
__device__ __forceinline__ v16bf load_frag_b(const unsigned short* ldsT, int stride,
                                             int n0, int k0, int lane) {
    union { v16bf v; uint4v q[2]; } f;
    const int n = n0 + (lane & 15);
    const int g = lane >> 4;
    const unsigned short* p = ldsT + n * stride + k0 + (g << 4);
    f.q[0] = *(const uint4v*)(p);
    f.q[1] = *(const uint4v*)(p + 8);
    return f.v;
}

__device__ __forceinline__ v8f wmma_bf16(v16bf a, v16bf b, v8f c) {
    return __builtin_amdgcn_wmma_f32_16x16x32_bf16(
        false, a, false, b, (short)0, c, false, false);
}

// ---------------------------------------------------------------------------
// Kernel 1a: h[j] = elu(dot(A.flatten, W1[j,:])), j in [0,256)
// ---------------------------------------------------------------------------
__global__ void gcn_h_kernel(const float* __restrict__ A,
                             const float* __restrict__ W1,
                             float* __restrict__ h_out) {
    __shared__ float red[256];
    const int j = blockIdx.x;
    const int tid = threadIdx.x;
    const float* w = W1 + (size_t)j * 4096;
    float acc = 0.f;
    for (int i = tid; i < 4096; i += 256) acc = fmaf(A[i], w[i], acc);
    red[tid] = acc;
    __syncthreads();
    for (int s = 128; s > 0; s >>= 1) {
        if (tid < s) red[tid] += red[tid + s];
        __syncthreads();
    }
    if (tid == 0) h_out[j] = elu1(red[0]);
}

// ---------------------------------------------------------------------------
// Kernel 1b: Ad = relu(h @ W2^T) [64x64]; row-normalize; emit bf16 D_invA.
// ---------------------------------------------------------------------------
__global__ void gcn_adj_kernel(const float* __restrict__ W2,
                               const float* __restrict__ h,
                               unsigned short* __restrict__ dinv_bf) {
    __shared__ float sh[256];
    __shared__ float row[64];
    const int r = blockIdx.x;
    const int c = threadIdx.x;
    for (int i = c; i < 256; i += 64) sh[i] = h[i];
    __syncthreads();
    const float* w = W2 + (size_t)(r * 64 + c) * 256;
    float acc = 0.f;
#pragma unroll 4
    for (int j = 0; j < 256; ++j) acc = fmaf(sh[j], w[j], acc);
    acc = acc > 0.f ? acc : 0.f;     // relu
    row[c] = acc;
    __syncthreads();
    float s = 0.f;
#pragma unroll
    for (int i = 0; i < 64; ++i) s += row[i];
    s = fmaxf(s, 1e-6f);
    dinv_bf[r * 64 + c] = f2bf(acc / s);
}

// ---------------------------------------------------------------------------
// Kernel 2: fused per-(band, batch-chunk) WMMA chain. 16 wave32.
// Wave w: fixed tm = w>>2, tn in {(w&3)*2, (w&3)*2+1}.
// ---------------------------------------------------------------------------
__global__ void gcn_main_kernel(const float* __restrict__ mu,
                                const float* __restrict__ T1,
                                const float* __restrict__ T2,
                                const unsigned short* __restrict__ dinv_bf,
                                float* __restrict__ out) {
    extern __shared__ unsigned short smem[];
    unsigned short* sT1 = smem;                    // 128 x ST_T  (T1 transposed)
    unsigned short* sT2 = sT1 + 128 * ST_T;        // 128 x ST_T  (T2 transposed)
    unsigned short* sD  = sT2 + 128 * ST_T;        // 64  x ST_D  (D_invA row-major)
    unsigned short* sMu = sD  + 64  * ST_D;        // 64  x ST_MU (mu tile row-major)
    unsigned short* sH1 = sMu + 64  * ST_MU;       // 128 x ST_H1 (h1 transposed)
    unsigned short* sH2 = sH1 + 128 * ST_H1;       // 64  x ST_H2 (h2 row-major)

    const int tid  = threadIdx.x;
    const int lane = tid & 31;
    const int wid  = tid >> 5;                            // 0..15
    const int gb   = blockIdx.x / (NBATCH / NB);          // band 0..4
    const int b0   = (blockIdx.x % (NBATCH / NB)) * NB;

    // ---- stage band-constant operands -------------------------------------
    const float* t1g = T1 + (size_t)gb * FDIM * FDIM;
    const float* t2g = T2 + (size_t)gb * FDIM * FDIM;
#pragma unroll
    for (int j = 0; j < 8; ++j) {                          // 4096 float4 / 512
        int i4 = tid + j * NTHR;
        int f  = i4 >> 5;                                  // row in T (f index)
        int h4 = (i4 & 31) << 2;                           // 4 consecutive h
        float4v v1 = *((const float4v*)t1g + i4);
        float4v v2 = *((const float4v*)t2g + i4);
        v4bf h1 = __builtin_convertvector(v1, v4bf);
        v4bf h2 = __builtin_convertvector(v2, v4bf);
        union { v4bf v; unsigned short s[4]; } u1, u2;
        u1.v = h1; u2.v = h2;
#pragma unroll
        for (int c = 0; c < 4; ++c) {
            sT1[(h4 + c) * ST_T + f] = u1.s[c];
            sT2[(h4 + c) * ST_T + f] = u2.s[c];
        }
    }
#pragma unroll
    for (int j = 0; j < 2; ++j) {                          // 1024 ushort4 / 512
        int i4 = tid + j * NTHR;
        int r  = i4 >> 4;
        int c4 = (i4 & 15) << 2;
        *(ushort4v*)(sD + r * ST_D + c4) = *((const ushort4v*)dinv_bf + i4);
    }
    __syncthreads();

    const int nloc = lane & 15;
    const int gsel = lane >> 4;
    const int tm   = wid >> 2;                             // fixed row-tile per wave
    const int tn0  = (wid & 3) << 1;                       // 2 col-tiles per wave

    // D_invA A-fragments are loop-invariant: pin in registers for whole block.
    v16bf aD[2];
#pragma unroll
    for (int k = 0; k < 2; ++k)
        aD[k] = load_frag_a(sD, ST_D, tm * 16, k * 32, lane);

    for (int bi = 0; bi < NB; ++bi) {
        const int b = b0 + bi;
        const float* mub = mu + ((size_t)(gb * NBATCH + b)) * (EDIM * FDIM);

        // ---- stage mu tile (f32 -> bf16, row-major), non-temporal ---------
#pragma unroll
        for (int j = 0; j < 4; ++j) {                      // 2048 float4 / 512
            int i4 = tid + j * NTHR;
            int e  = i4 >> 5;
            int f4 = (i4 & 31) << 2;
            float4v v = __builtin_nontemporal_load((const float4v*)mub + i4);
            v4bf hv = __builtin_convertvector(v, v4bf);
            union { v4bf v; ushort4v s; } uu; uu.v = hv;
            *(ushort4v*)(sMu + e * ST_MU + f4) = uu.s;
        }
        // prefetch next batch tile while we compute this one
        if (bi + 1 < NB) {
            const float* munext = mub + EDIM * FDIM;
            __builtin_prefetch(munext + (tid << 4), 0, 0); // 512 x 64B = 32KB
        }
        __syncthreads();

        // ---- M1: h1 = elu(mu @ T1) -> sH1 (transposed [h][e]) -------------
        {
            v16bf aMu[4];
#pragma unroll
            for (int k = 0; k < 4; ++k)
                aMu[k] = load_frag_a(sMu, ST_MU, tm * 16, k * 32, lane);
#pragma unroll
            for (int t = 0; t < 2; ++t) {
                const int tn = tn0 + t;
                v8f acc = {};
#pragma unroll
                for (int k = 0; k < 4; ++k)
                    acc = wmma_bf16(aMu[k], load_frag_b(sT1, ST_T, tn * 16, k * 32, lane), acc);
#pragma unroll
                for (int v = 0; v < 8; ++v) acc[v] = elu1(acc[v]);
                v8bf hb = __builtin_convertvector(acc, v8bf);
                union { v8bf v; unsigned short s[8]; } uh; uh.v = hb;
                const int n = tn * 16 + nloc;
#pragma unroll
                for (int v = 0; v < 8; ++v) {
                    int e = tm * 16 + v + (gsel << 3);
                    sH1[n * ST_H1 + e] = uh.s[v];
                }
            }
        }
        __syncthreads();

        // ---- M2: h2 = D_invA @ h1 -> sH2 (row-major [x][h]) ---------------
#pragma unroll
        for (int t = 0; t < 2; ++t) {
            const int tn = tn0 + t;
            v8f acc = {};
#pragma unroll
            for (int k = 0; k < 2; ++k)                    // K = 64
                acc = wmma_bf16(aD[k], load_frag_b(sH1, ST_H1, tn * 16, k * 32, lane), acc);
            v8bf hb = __builtin_convertvector(acc, v8bf);
            union { v8bf v; unsigned short s[8]; } uh; uh.v = hb;
            const int n = tn * 16 + nloc;
#pragma unroll
            for (int v = 0; v < 8; ++v) {
                int x = tm * 16 + v + (gsel << 3);
                sH2[x * ST_H2 + n] = uh.s[v];
            }
        }
        __syncthreads();

        // ---- M3: out = elu(h2 @ T2) -> global (non-temporal) --------------
        {
            float* outb = out + ((size_t)(gb * NBATCH + b)) * (EDIM * FDIM);
            v16bf aH2[4];
#pragma unroll
            for (int k = 0; k < 4; ++k)
                aH2[k] = load_frag_a(sH2, ST_H2, tm * 16, k * 32, lane);
#pragma unroll
            for (int t = 0; t < 2; ++t) {
                const int tn = tn0 + t;
                v8f acc = {};
#pragma unroll
                for (int k = 0; k < 4; ++k)
                    acc = wmma_bf16(aH2[k], load_frag_b(sT2, ST_T, tn * 16, k * 32, lane), acc);
                const int n = tn * 16 + nloc;
#pragma unroll
                for (int v = 0; v < 8; ++v) {
                    int e = tm * 16 + v + (gsel << 3);
                    __builtin_nontemporal_store(elu1(acc[v]), &outb[e * FDIM + n]);
                }
            }
        }
        // next iteration's staging barrier fences sH2 readers vs new writers
    }
}

// ---------------------------------------------------------------------------
extern "C" void kernel_launch(void* const* d_in, const int* in_sizes, int n_in,
                              void* d_out, int out_size, void* d_ws, size_t ws_size,
                              hipStream_t stream) {
    const float* mu = (const float*)d_in[0];
    const float* A  = (const float*)d_in[1];
    const float* W1 = (const float*)d_in[2];
    const float* W2 = (const float*)d_in[3];
    const float* T1 = (const float*)d_in[4];
    const float* T2 = (const float*)d_in[5];
    float* out = (float*)d_out;

    unsigned short* dinv_bf = (unsigned short*)d_ws;                 // 4096 * 2B
    float*          hbuf    = (float*)((char*)d_ws + 8192);          // 256  * 4B

    // squeeze-excite dynamic adjacency (tiny)
    gcn_h_kernel<<<dim3(256), dim3(256), 0, stream>>>(A, W1, hbuf);
    gcn_adj_kernel<<<dim3(64), dim3(64), 0, stream>>>(W2, hbuf, dinv_bf);

    // fused WMMA chain: 5 bands x 128 chunks of 8 batch items, 16 waves each
    gcn_main_kernel<<<dim3(NBANDS * (NBATCH / NB)), dim3(NTHR), LDS_BYTES, stream>>>(
        mu, T1, T2, dinv_bf, out);
}
